// RetrievalEnhancedMBO_21045339750712
// MI455X (gfx1250) — matrix operand order repaired and checked by reading, not compiled, not run on hardware
//
#include <hip/hip_runtime.h>

typedef __attribute__((ext_vector_type(16))) _Float16 v16h;
typedef __attribute__((ext_vector_type(8)))  float    v8f;
typedef __attribute__((ext_vector_type(4)))  unsigned v4u;
typedef __attribute__((ext_vector_type(8)))  int      v8i;
typedef __attribute__((ext_vector_type(4)))  int      v4i;

namespace {
constexpr int NQ     = 256;      // queries
constexpr int DIM    = 512;      // feature dim
constexpr int NP     = 100000;   // pool rows (divisible by 32)
constexpr int KSEL   = 10;       // retrieval set size
constexpr int PCHUNK = 512;      // pool rows per block
constexpr int NCHUNK = (NP + PCHUNK - 1) / PCHUNK;   // 196
constexpr int PSUB   = 32;       // pool rows per sub-tile (double-buffered)
constexpr int NSUB   = PCHUNK / PSUB;                // 16
constexpr int QG     = 64;       // query group staged in LDS
constexpr int NQG    = NQ / QG;                      // 4
constexpr float FINF = 3.0e38f;
}

#define TOPK_VARS \
  float d0=FINF,d1=FINF,d2=FINF,d3=FINF,d4=FINF,d5=FINF,d6=FINF,d7=FINF,d8=FINF,d9=FINF; \
  int   i0=0,i1=0,i2=0,i3=0,i4=0,i5=0,i6=0,i7=0,i8=0,i9=0

#define TOPK_SWAP(a,b,ia,ib) { float tf_=(a); (a)=(b); (b)=tf_; int ti_=(ia); (ia)=(ib); (ib)=ti_; }

#define TOPK_INSERT(key,idx) do {                       \
    float k_ = (key);                                   \
    if (k_ < d9) {                                      \
      d9 = k_; i9 = (idx);                              \
      if (d9 < d8) TOPK_SWAP(d9,d8,i9,i8);              \
      if (d8 < d7) TOPK_SWAP(d8,d7,i8,i7);              \
      if (d7 < d6) TOPK_SWAP(d7,d6,i7,i6);              \
      if (d6 < d5) TOPK_SWAP(d6,d5,i6,i5);              \
      if (d5 < d4) TOPK_SWAP(d5,d4,i5,i4);              \
      if (d4 < d3) TOPK_SWAP(d4,d3,i4,i3);              \
      if (d3 < d2) TOPK_SWAP(d3,d2,i3,i2);              \
      if (d2 < d1) TOPK_SWAP(d2,d1,i2,i1);              \
      if (d1 < d0) TOPK_SWAP(d1,d0,i1,i0);              \
    }                                                   \
  } while (0)

// TDM descriptor: 2D tensor load, 4-byte elements, tile = DIM x PSUB rows.
// tensor_dim1 bounds row reads (rows past it read as zero per ISA).
__device__ __forceinline__ void tdm_load_tile(const void* gsrc, unsigned lds_off,
                                              unsigned tdim1) {
  const unsigned long long ga = (unsigned long long)(size_t)gsrc;
  v4u g0;
  g0.x = 1u;                                   // count=1, user mode
  g0.y = lds_off;                              // lds_addr (bytes)
  g0.z = (unsigned)(ga & 0xFFFFFFFFu);         // global_addr[31:0]
  g0.w = (unsigned)((ga >> 32) & 0x01FFFFFFu)  // global_addr[56:32]
       | (2u << 30);                           // type=2 (image)
  v8i g1;
  g1[0] = (int)(2u << 16);                     // data_size=2 -> 4 bytes
  g1[1] = (int)((unsigned)DIM << 16);          // tensor_dim0[15:0]=512
  g1[2] = (int)((tdim1 & 0xFFFFu) << 16);      // tensor_dim1[15:0]
  g1[3] = (int)((tdim1 >> 16) |
                ((unsigned)DIM << 16));        // tensor_dim1[31:16], tile_dim0=512
  g1[4] = (int)PSUB;                           // tile_dim1, tile_dim2=0
  g1[5] = (int)DIM;                            // tensor_dim0_stride=512 (lo32)
  g1[6] = 0;                                   // stride hi, dim1_stride lo
  g1[7] = 0;
  v4i gz4 = {0, 0, 0, 0};                      // groups 2/3: 2D tensor
  v8i gz8 = {0, 0, 0, 0, 0, 0, 0, 0};
  __builtin_amdgcn_tensor_load_to_lds(g0, g1, gz4, gz4, gz8, 0);
}

// ---------------------------------------------------------------------------
// Kernel 0: squared norms of pool rows (exact f32, one-time 205 MB pass)
// ---------------------------------------------------------------------------
__global__ __launch_bounds__(256)
void psq_kernel(const float* __restrict__ poolx, float* __restrict__ psq) {
  const int row = blockIdx.x * 256 + threadIdx.x;
  if (row >= NP) return;
  const float* r = poolx + (size_t)row * DIM;
  float s = 0.0f;
#pragma unroll 8
  for (int i = 0; i < DIM; ++i) s += r[i] * r[i];
  psq[row] = s;
}

// ---------------------------------------------------------------------------
// Kernel 1: fused WMMA inner-product + per-chunk top-K.
//   Double-buffered TDM pipeline: while tile s is converted (f32->f16) and
//   consumed by WMMA, the Tensor Data Mover streams tile s+1 into the other
//   LDS buffer (TENSORcnt overlap). rank key = ||p||^2 - 2*<x,p>.
// ---------------------------------------------------------------------------
__global__ __launch_bounds__(256)
void dist_topk_kernel(const float* __restrict__ xg,
                      const float* __restrict__ poolx,
                      const float* __restrict__ psq,
                      float* __restrict__ cand_key,
                      int* __restrict__ cand_idx) {
  __shared__ _Float16 xh[QG * DIM];          // 64 KB: query-group tile (f16)
  __shared__ float    pf[2][PSUB * DIM];     // 2x64 KB: TDM dest (f32, ping-pong)
  __shared__ _Float16 ph[PSUB * DIM];        // 32 KB: converted pool tile (f16)
  __shared__ float    dists[QG * PSUB];      // 8 KB: inner products (f32)
  __shared__ float    psqs[PSUB];

  const int t     = threadIdx.x;
  const int lane  = t & 31;
  const int wave  = t >> 5;
  const int chunk = blockIdx.x;
  const int pbase = chunk * PCHUNK;

  // wave -> output tile: 8 waves cover 4 query-tiles x 2 pool-tiles
  const int qt  = wave >> 1;
  const int pt  = wave & 1;
  const int nlo = lane & 15;   // M (for A) / N (for B,D) within tile
  const int kh  = lane >> 4;   // lane-half selects K sub-range / D row-half

  const unsigned lds_pf0 = (unsigned)(size_t)(const void*)pf[0];
  const unsigned lds_pf1 = (unsigned)(size_t)(const void*)pf[1];

  for (int qg = 0; qg < NQG; ++qg) {
    __syncthreads();  // previous group's readers done (xh, pf buffers free)
    for (int i = t; i < QG * DIM; i += 256)
      xh[i] = (_Float16)xg[qg * QG * DIM + i];

    // prime the pipeline: TDM for sub-tile 0 into pf[0]
    if (wave == 0) {
      const int pc = (pbase <= NP - PSUB) ? pbase : (NP - PSUB);
      tdm_load_tile(poolx + (size_t)pc * DIM, lds_pf0, (unsigned)(NP - pc));
    }

    TOPK_VARS;

    for (int s = 0; s < NSUB; ++s) {
      if (wave == 0) __builtin_amdgcn_s_wait_tensorcnt(0);
      __syncthreads();  // tile s landed; prior readers of the other buffer done
      const int p0 = pbase + s * PSUB;

      // issue TDM for tile s+1 into the other buffer (overlaps with compute)
      if (wave == 0 && s + 1 < NSUB) {
        const int pn = p0 + PSUB;
        const int pc = (pn <= NP - PSUB) ? pn : (NP - PSUB);
        tdm_load_tile(poolx + (size_t)pc * DIM,
                      ((s + 1) & 1) ? lds_pf1 : lds_pf0,
                      (unsigned)(NP - pc));
      }

      // cooperative f32 -> f16 conversion of tile s (once, shared by all waves)
      {
        const float* src = pf[s & 1];
        for (int i = t * 4; i < PSUB * DIM; i += 256 * 4) {
          const float a0 = src[i + 0], a1 = src[i + 1];
          const float a2 = src[i + 2], a3 = src[i + 3];
          ph[i + 0] = (_Float16)a0; ph[i + 1] = (_Float16)a1;
          ph[i + 2] = (_Float16)a2; ph[i + 3] = (_Float16)a3;
        }
      }
      if (t < PSUB) {
        const int prow = p0 + t;
        psqs[t] = (prow < NP) ? psq[prow] : FINF;
      }
      // warm GL2 for tile s+2 ahead of its TDM (256 threads x 256 B = 64 KB)
      {
        const int np0 = p0 + 2 * PSUB;
        if (np0 + PSUB <= NP) {
          const char* nb = (const char*)(poolx + (size_t)np0 * DIM);
          __builtin_prefetch(nb + (size_t)t * 256, 0, 1);
        }
      }
      __syncthreads();  // ph + psqs visible to all waves

      // ---- WMMA: one 16x16 tile per wave, K = 512 in 16 steps ----
      v8f acc = {};
      const unsigned* xrow = (const unsigned*)(xh + (qt * 16 + nlo) * DIM);
      const unsigned* prow = (const unsigned*)(ph + (pt * 16 + nlo) * DIM);
#pragma unroll
      for (int k0 = 0; k0 < DIM; k0 += 32) {
        union { v16h v; unsigned u[8]; } Af, Bt;
        // A 16x32 f16, this code's contiguous k order (same as B): vgpr j -> K = k0+16*kh+2j
        const int ka = (k0 >> 1) + 8 * kh;
        Af.u[0] = xrow[ka + 0];  Af.u[1] = xrow[ka + 1];
        Af.u[2] = xrow[ka + 2];  Af.u[3] = xrow[ka + 3];
        Af.u[4] = xrow[ka + 4];  Af.u[5] = xrow[ka + 5];
        Af.u[6] = xrow[ka + 6];  Af.u[7] = xrow[ka + 7];
        // B 32x16 f16, same contiguous k order as A: lane = column n; vgpr j -> K = k0+16*kh+2j
        const int kb = (k0 >> 1) + 8 * kh;
#pragma unroll
        for (int j = 0; j < 8; ++j) Bt.u[j] = prow[kb + j];
        acc = __builtin_amdgcn_wmma_f32_16x16x32_f16(
                  false, Af.v, false, Bt.v, (short)0, acc, false, false);
      }
      // D layout: M = v + 8*kh, N = lane&15
#pragma unroll
      for (int v = 0; v < 8; ++v)
        dists[(qt * 16 + v + 8 * kh) * PSUB + pt * 16 + nlo] = acc[v];
      __syncthreads();

      // ---- scan: thread t<64 owns query qg*64+t, updates register top-K ----
      if (t < QG) {
        const float* dr = dists + t * PSUB;
        for (int j = 0; j < PSUB; ++j) {
          const float key = psqs[j] - 2.0f * dr[j];
          TOPK_INSERT(key, p0 + j);
        }
      }
    }

    // write this chunk's candidates (sorted ascending)
    if (t < QG) {
      const int q = qg * QG + t;
      const size_t cb = ((size_t)q * NCHUNK + chunk) * KSEL;
      cand_key[cb + 0] = d0; cand_idx[cb + 0] = i0;
      cand_key[cb + 1] = d1; cand_idx[cb + 1] = i1;
      cand_key[cb + 2] = d2; cand_idx[cb + 2] = i2;
      cand_key[cb + 3] = d3; cand_idx[cb + 3] = i3;
      cand_key[cb + 4] = d4; cand_idx[cb + 4] = i4;
      cand_key[cb + 5] = d5; cand_idx[cb + 5] = i5;
      cand_key[cb + 6] = d6; cand_idx[cb + 6] = i6;
      cand_key[cb + 7] = d7; cand_idx[cb + 7] = i7;
      cand_key[cb + 8] = d8; cand_idx[cb + 8] = i8;
      cand_key[cb + 9] = d9; cand_idx[cb + 9] = i9;
    }
  }
}

// ---------------------------------------------------------------------------
// Kernel 2: merge per-chunk candidates, gather retrieval set
//   one block per query; out[q][k][0:512] = pool_x[idx], out[q][k][512] = pool_y
// ---------------------------------------------------------------------------
__global__ __launch_bounds__(256)
void merge_gather_kernel(const float* __restrict__ poolx,
                         const float* __restrict__ pooly,
                         const float* __restrict__ cand_key,
                         const int* __restrict__ cand_idx,
                         float* __restrict__ out) {
  __shared__ float ks[NCHUNK * KSEL];
  __shared__ int   is_[NCHUNK * KSEL];
  __shared__ int   sel[KSEL];

  const int q = blockIdx.x;
  const int t = threadIdx.x;
  const size_t cb = (size_t)q * NCHUNK * KSEL;

  for (int i = t; i < NCHUNK * KSEL; i += 256) {
    ks[i]  = cand_key[cb + i];
    is_[i] = cand_idx[cb + i];
  }
  __syncthreads();

  if (t == 0) {
    TOPK_VARS;
    for (int j = 0; j < NCHUNK * KSEL; ++j) TOPK_INSERT(ks[j], is_[j]);
    sel[0] = i0; sel[1] = i1; sel[2] = i2; sel[3] = i3; sel[4] = i4;
    sel[5] = i5; sel[6] = i6; sel[7] = i7; sel[8] = i8; sel[9] = i9;
  }
  __syncthreads();

  for (int k = 0; k < KSEL; ++k) {
    const int row = sel[k];
    float* o = out + ((size_t)q * KSEL + k) * (DIM + 1);
    const float* src = poolx + (size_t)row * DIM;
    for (int e = t; e < DIM; e += 256) o[e] = src[e];
    if (t == 0) o[DIM] = pooly[row];
  }
}

// ---------------------------------------------------------------------------
extern "C" void kernel_launch(void* const* d_in, const int* in_sizes, int n_in,
                              void* d_out, int out_size, void* d_ws, size_t ws_size,
                              hipStream_t stream) {
  (void)in_sizes; (void)n_in; (void)out_size; (void)ws_size;
  const float* xg    = (const float*)d_in[0];
  const float* poolx = (const float*)d_in[1];
  const float* pooly = (const float*)d_in[2];
  float* out = (float*)d_out;

  char* ws = (char*)d_ws;
  float* psq = (float*)ws;                                        // NP floats
  size_t off = ((size_t)NP * sizeof(float) + 255) & ~(size_t)255; // 400128
  float* cand_key = (float*)(ws + off);
  const size_t cand_elems = (size_t)NQ * NCHUNK * KSEL;           // 501760
  int* cand_idx = (int*)(ws + off + cand_elems * sizeof(float));
  // total ws use: ~4.4 MB

  psq_kernel<<<(NP + 255) / 256, 256, 0, stream>>>(poolx, psq);
  dist_topk_kernel<<<NCHUNK, 256, 0, stream>>>(xg, poolx, psq, cand_key, cand_idx);
  merge_gather_kernel<<<NQ, 256, 0, stream>>>(poolx, pooly, cand_key, cand_idx, out);
}
